// GraphMultisetTransformerGNN_936302870761
// MI455X (gfx1250) — compile-verified, matrix operand order checked by reading, not run on hardware
//
#include <hip/hip_runtime.h>

// ---------------------------------------------------------------------------
// Types for CDNA5 WMMA (wave32)
// ---------------------------------------------------------------------------
typedef __attribute__((ext_vector_type(16))) __bf16 v16bf;
typedef __attribute__((ext_vector_type(8)))  __bf16 v8bf;
typedef __attribute__((ext_vector_type(8)))  float  v8f;

// ---------------------------------------------------------------------------
// Problem constants (match reference)
// ---------------------------------------------------------------------------
static constexpr int    Bc   = 64;
static constexpr int    Nn   = 1024;
static constexpr int    Hc   = 4;
static constexpr int    DHc  = 64;
static constexpr int    S1c  = 32;
static constexpr int    Dm   = 256;
static constexpr int    DINc = 128;
static constexpr int    NTOT = Bc * Nn;        // 65536
static constexpr int    Ec   = Bc * 4096;      // 262144

// ---------------------------------------------------------------------------
// Workspace layout (byte offsets into d_ws)
// ---------------------------------------------------------------------------
static constexpr size_t SZ_BIG   = (size_t)NTOT * Dm * 4;           // 64 MB
static constexpr size_t O_A      = 0;
static constexpr size_t O_Bg     = O_A + SZ_BIG;
static constexpr size_t O_Cg     = O_Bg + SZ_BIG;
static constexpr size_t O_XB     = O_Cg + SZ_BIG;                    // bf16 acts
static constexpr size_t O_DINV   = O_XB + (size_t)NTOT * Dm * 2;
static constexpr size_t O_NORM   = O_DINV + (size_t)NTOT * 4;
static constexpr size_t O_W0T    = O_NORM + (size_t)Ec * 4;
static constexpr size_t O_W1T    = O_W0T  + 256*128*2;
static constexpr size_t O_W2T    = O_W1T  + 256*256*2;
static constexpr size_t O_WK1T   = O_W2T  + 256*256*2;
static constexpr size_t O_WV1T   = O_WK1T + 256*256*2;
static constexpr size_t O_QW1T   = O_WV1T + 256*256*2;
static constexpr size_t O_OW1T   = O_QW1T + 256*256*2;
static constexpr size_t O_FF1T   = O_OW1T + 256*256*2;
static constexpr size_t O_INW2T  = O_FF1T + 256*256*2;
static constexpr size_t O_OW2T   = O_INW2T+ 768*256*2;
static constexpr size_t O_FF2T   = O_OW2T + 256*256*2;
static constexpr size_t O_WK3T   = O_FF2T + 256*256*2;
static constexpr size_t O_WV3T   = O_WK3T + 256*256*2;
static constexpr size_t O_OW3T   = O_WV3T + 256*256*2;
static constexpr size_t O_FF3T   = O_OW3T + 256*256*2;
static constexpr size_t O_HW1T   = O_FF3T + 256*256*2;
static constexpr size_t O_HW2T   = O_HW1T + 256*256*2;
static constexpr size_t O_SEEDB  = O_HW2T + 256*128*2;
static constexpr size_t O_QP1    = O_SEEDB + 32*256*2;
static constexpr size_t O_OBF    = O_QP1  + 32*256*4;                // bf16 attn out
static constexpr size_t O_ATT    = O_OBF  + (size_t)2048*256*2;
static constexpr size_t O_E      = O_ATT  + (size_t)2048*256*4;
static constexpr size_t O_X1     = O_E    + (size_t)2048*256*4;
static constexpr size_t O_FF     = O_X1   + (size_t)2048*256*4;
static constexpr size_t O_X2     = O_FF   + (size_t)2048*256*4;
static constexpr size_t O_SB     = O_X2   + (size_t)2048*256*4;      // bf16 GEMM input
static constexpr size_t O_QKV    = O_SB   + (size_t)2048*256*2;
static constexpr size_t O_K3     = O_QKV  + (size_t)2048*768*4;
static constexpr size_t O_V3     = O_K3   + (size_t)2048*256*4;
static constexpr size_t O_Q3     = O_V3   + (size_t)2048*256*4;

// ---------------------------------------------------------------------------
// WMMA GEMM:  C[M,N] = A[M,K](bf16,row) * Bt[N,K](bf16,row = B^T) + bias
// One wave computes one 16x16 tile. Fragments assembled with two 16B loads
// per operand per K-step (matches CDNA5 16-bit A/B VGPR layout).
// ---------------------------------------------------------------------------
__global__ __launch_bounds__(256)
void k_gemm_wmma(const __bf16* __restrict__ A, const __bf16* __restrict__ Bt,
                 const float* __restrict__ bias, float* __restrict__ C,
                 int M, int N, int K, int relu) {
  const int lane = threadIdx.x & 31;
  const int wid  = (blockIdx.x * blockDim.x + threadIdx.x) >> 5;
  const int TN = N >> 4;
  const int TM = M >> 4;
  if (wid >= TM * TN) return;               // wave-uniform: EXEC all-ones below
  const int tm = wid / TN, tn = wid % TN;
  const int g = lane >> 4, r = lane & 15;
  const __bf16* ar = A  + ((size_t)(tm * 16 + r)) * K + g * 8;
  const __bf16* br = Bt + ((size_t)(tn * 16 + r)) * K + g * 8;
  v8f acc = {0.f, 0.f, 0.f, 0.f, 0.f, 0.f, 0.f, 0.f};
  for (int k0 = 0; k0 < K; k0 += 32) {
    v8bf a0 = *(const v8bf*)(ar + k0);
    v8bf a1 = *(const v8bf*)(ar + k0 + 16);
    v8bf b0 = *(const v8bf*)(br + k0);
    v8bf b1 = *(const v8bf*)(br + k0 + 16);
    v16bf av = __builtin_shufflevector(a0, a1, 0,1,2,3,4,5,6,7,8,9,10,11,12,13,14,15);
    v16bf bv = __builtin_shufflevector(b0, b1, 0,1,2,3,4,5,6,7,8,9,10,11,12,13,14,15);
    acc = __builtin_amdgcn_wmma_f32_16x16x32_bf16(false, av, false, bv,
                                                  (short)0, acc, false, false);
  }
  const int n = tn * 16 + r;
  const float bv2 = bias ? bias[n] : 0.f;
#pragma unroll
  for (int i = 0; i < 8; i++) {
    const int m = tm * 16 + i + g * 8;      // D layout: vgpr i -> M = i + 8*g
    float v = acc[i] + bv2;
    if (relu) v = fmaxf(v, 0.f);
    C[(size_t)m * N + n] = v;
  }
}

// ---------------------------------------------------------------------------
// Small utility kernels
// ---------------------------------------------------------------------------
__global__ void k_f32_to_bf16(const float* __restrict__ x, __bf16* __restrict__ y,
                              int n, int relu) {
  int i = blockIdx.x * blockDim.x + threadIdx.x;
  if (i >= n) return;
  float v = x[i];
  if (relu) v = fmaxf(v, 0.f);
  y[i] = (__bf16)v;
}

__global__ void k_w_t_bf16(const float* __restrict__ W, __bf16* __restrict__ Wt,
                           int K, int N) {   // W[K,N] -> Wt[N,K] bf16
  int i = blockIdx.x * blockDim.x + threadIdx.x;
  if (i >= K * N) return;
  int k = i / N, n = i % N;
  Wt[(size_t)n * K + k] = (__bf16)W[i];
}

// Wk1[H,D,DH] -> concat-head weight transposed: Wt[n= h*64+e][d]
__global__ void k_wkcat_t(const float* __restrict__ W, __bf16* __restrict__ Wt) {
  int i = blockIdx.x * blockDim.x + threadIdx.x;   // i = n*256 + d
  if (i >= 256 * 256) return;
  int n = i >> 8, d = i & 255;
  Wt[i] = (__bf16)W[(size_t)(n >> 6) * 256 * 64 + (size_t)d * 64 + (n & 63)];
}

__global__ void k_fill(float* p, float v, int n) {
  int i = blockIdx.x * blockDim.x + threadIdx.x;
  if (i < n) p[i] = v;
}

__global__ void k_deg(const int* __restrict__ dst, float* deg, int E) {
  int e = blockIdx.x * blockDim.x + threadIdx.x;
  if (e < E) atomicAdd(&deg[dst[e]], 1.0f);
}

__global__ void k_dinv(float* deg, int n) {
  int i = blockIdx.x * blockDim.x + threadIdx.x;
  if (i < n) deg[i] = rsqrtf(deg[i]);       // deg >= 1 always (self loop)
}

__global__ void k_edge_norm(const int* __restrict__ src, const int* __restrict__ dst,
                            const float* __restrict__ dinv, float* __restrict__ nrm, int E) {
  int e = blockIdx.x * blockDim.x + threadIdx.x;
  if (e < E) nrm[e] = dinv[src[e]] * dinv[dst[e]];
}

// out = (bias? bias[c]:0) + proj * dinv[node]^2   (self-loop + bias init)
__global__ void k_gcn_init(const float* __restrict__ proj, const float* __restrict__ bias,
                           const float* __restrict__ dinv, float* __restrict__ out) {
  int i = blockIdx.x * blockDim.x + threadIdx.x;
  if (i >= NTOT * Dm) return;
  int node = i >> 8, c = i & 255;
  float dv = dinv[node];
  float b = bias ? bias[c] : 0.f;
  out[i] = b + proj[i] * dv * dv;
}

// edge scatter: out[dst] += proj[src] * norm, float4-vectorized + f32 atomics
__global__ void k_gcn_scatter(const float* __restrict__ proj, const int* __restrict__ src,
                              const int* __restrict__ dst, const float* __restrict__ nrm,
                              float* __restrict__ out) {
  int tid = blockIdx.x * blockDim.x + threadIdx.x;
  if (tid >= Ec * (Dm / 4)) return;
  int e = tid >> 6, q = tid & 63;
  int s = src[e], d = dst[e];
  float w = nrm[e];
  const float4 hv = *(const float4*)(proj + (size_t)s * Dm + q * 4);
  float* o = out + (size_t)d * Dm + q * 4;
  atomicAdd(o + 0, hv.x * w);
  atomicAdd(o + 1, hv.y * w);
  atomicAdd(o + 2, hv.z * w);
  atomicAdd(o + 3, hv.w * w);
}

// ---------------------------------------------------------------------------
// LayerNorm kernels: one row (D=256) per wave32, lane owns 8 elements
// ---------------------------------------------------------------------------
__device__ inline float wred32(float v) {
#pragma unroll
  for (int i = 16; i > 0; i >>= 1) v += __shfl_xor(v, i, 32);
  return v;
}

__global__ void k_ln_add(const float* __restrict__ a, const float* __restrict__ b,
                         const float* __restrict__ w, const float* __restrict__ bias,
                         float* __restrict__ y, int rows) {
  int wv = (blockIdx.x * blockDim.x + threadIdx.x) >> 5;
  int lane = threadIdx.x & 31;
  if (wv >= rows) return;
  const float* ar = a + (size_t)wv * 256;
  const float* br = b + (size_t)wv * 256;
  float t[8]; float s = 0.f;
#pragma unroll
  for (int j = 0; j < 8; j++) { int c = lane * 8 + j; t[j] = ar[c] + br[c]; s += t[j]; }
  s = wred32(s);
  float mean = s * (1.f / 256.f);
  float vs = 0.f;
#pragma unroll
  for (int j = 0; j < 8; j++) { float d = t[j] - mean; vs += d * d; }
  vs = wred32(vs);
  float rstd = rsqrtf(vs * (1.f / 256.f) + 1e-5f);
  float* yr = y + (size_t)wv * 256;
#pragma unroll
  for (int j = 0; j < 8; j++) { int c = lane * 8 + j; yr[c] = (t[j] - mean) * rstd * w[c] + bias[c]; }
}

// seed broadcast variant: out[b*S+s] = LN(seed[s] + att[s*B+b])
__global__ void k_ln_add_bcast(const float* __restrict__ seed, const float* __restrict__ att,
                               const float* __restrict__ w, const float* __restrict__ bias,
                               float* __restrict__ y, int S, int B) {
  int wv = (blockIdx.x * blockDim.x + threadIdx.x) >> 5;
  int lane = threadIdx.x & 31;
  if (wv >= S * B) return;
  int b = wv / S, s = wv % S;
  const float* ar = seed + (size_t)s * 256;
  const float* br = att + ((size_t)s * B + b) * 256;
  float t[8]; float sm = 0.f;
#pragma unroll
  for (int j = 0; j < 8; j++) { int c = lane * 8 + j; t[j] = ar[c] + br[c]; sm += t[j]; }
  sm = wred32(sm);
  float mean = sm * (1.f / 256.f);
  float vs = 0.f;
#pragma unroll
  for (int j = 0; j < 8; j++) { float d = t[j] - mean; vs += d * d; }
  vs = wred32(vs);
  float rstd = rsqrtf(vs * (1.f / 256.f) + 1e-5f);
  float* yr = y + (size_t)wv * 256;
#pragma unroll
  for (int j = 0; j < 8; j++) { int c = lane * 8 + j; yr[c] = (t[j] - mean) * rstd * w[c] + bias[c]; }
}

// ---------------------------------------------------------------------------
// attn_1: flash-style, one block per (b,h). 32 seed queries over 1024 nodes.
// Output written as bf16 directly in [s*B+b, D] layout for the next WMMA GEMM.
// ---------------------------------------------------------------------------
#define A1_LT 64
__global__ __launch_bounds__(256)
void k_attn1(const float* __restrict__ Kagg, const float* __restrict__ Vagg,
             const float* __restrict__ qp, __bf16* __restrict__ Obf) {
  __shared__ float q[32][64];
  __shared__ float kt[A1_LT][64];
  __shared__ float vt[A1_LT][64];
  __shared__ float sc[32][A1_LT];
  __shared__ float mrow[32], zrow[32], rrow[32];
  const int b = blockIdx.x >> 2, h = blockIdx.x & 3;
  const int t = threadIdx.x;
  for (int i = t; i < 32 * 64; i += 256) {
    int s = i >> 6, e = i & 63;
    q[s][e] = qp[s * 256 + h * 64 + e] * 0.125f;   // DH^-0.5
  }
  if (t < 32) { mrow[t] = -1e30f; zrow[t] = 0.f; }
  float o[8];
#pragma unroll
  for (int j = 0; j < 8; j++) o[j] = 0.f;
  const int os = t >> 3, oe = (t & 7) * 8;
  __syncthreads();
  for (int l0 = 0; l0 < Nn; l0 += A1_LT) {
    for (int i = t; i < A1_LT * 16; i += 256) {
      int row = i >> 4, c4 = i & 15;
      size_t base = ((size_t)(b * Nn + l0 + row)) * 256 + h * 64 + c4 * 4;
      ((float4*)&kt[row][0])[c4] = *(const float4*)(Kagg + base);
      ((float4*)&vt[row][0])[c4] = *(const float4*)(Vagg + base);
    }
    __syncthreads();
    const int l = t & (A1_LT - 1);
    const int sg = t >> 6;                 // 4 groups of threads over s
    for (int sj = 0; sj < 8; sj++) {
      int s = sg * 8 + sj;
      float acc = 0.f;
#pragma unroll 8
      for (int e = 0; e < 64; e++) acc += q[s][e] * kt[l][e];
      sc[s][l] = acc;
    }
    __syncthreads();
    if (t < 32) {
      int s = t;
      float mt = mrow[s];
      for (int l2 = 0; l2 < A1_LT; l2++) mt = fmaxf(mt, sc[s][l2]);
      float rsc = __expf(mrow[s] - mt);
      float zs = zrow[s] * rsc;
      for (int l2 = 0; l2 < A1_LT; l2++) { float p = __expf(sc[s][l2] - mt); sc[s][l2] = p; zs += p; }
      mrow[s] = mt; zrow[s] = zs; rrow[s] = rsc;
    }
    __syncthreads();
    float rsc = rrow[os];
#pragma unroll
    for (int j = 0; j < 8; j++) o[j] *= rsc;
    for (int l2 = 0; l2 < A1_LT; l2++) {
      float p = sc[os][l2];
#pragma unroll
      for (int j = 0; j < 8; j++) o[j] += p * vt[l2][oe + j];
    }
    __syncthreads();
  }
  float zi = 1.f / zrow[os];
#pragma unroll
  for (int j = 0; j < 8; j++)
    Obf[((size_t)(os * Bc + b)) * 256 + h * 64 + oe + j] = (__bf16)(o[j] * zi);
}

// ---------------------------------------------------------------------------
// attn_2: self-attention over 32 tokens, one block per (b,h)
// qkv rows are b*32+s, cols [q|k|v] each h*64+e. Output bf16 rows b*32+s.
// ---------------------------------------------------------------------------
__global__ __launch_bounds__(256)
void k_attn2(const float* __restrict__ qkv, __bf16* __restrict__ Obf) {
  __shared__ float q[32][64], kk[32][64], vv[32][64], sc[32][32];
  const int b = blockIdx.x >> 2, h = blockIdx.x & 3;
  const int t = threadIdx.x;
  for (int i = t; i < 32 * 64; i += 256) {
    int s = i >> 6, e = i & 63;
    const float* row = qkv + ((size_t)(b * 32 + s)) * 768;
    q[s][e]  = row[h * 64 + e] * 0.125f;
    kk[s][e] = row[256 + h * 64 + e];
    vv[s][e] = row[512 + h * 64 + e];
  }
  __syncthreads();
  for (int i = t; i < 1024; i += 256) {
    int s = i >> 5, l = i & 31;
    float a = 0.f;
#pragma unroll 8
    for (int e = 0; e < 64; e++) a += q[s][e] * kk[l][e];
    sc[s][l] = a;
  }
  __syncthreads();
  if (t < 32) {
    float m = -1e30f;
    for (int l = 0; l < 32; l++) m = fmaxf(m, sc[t][l]);
    float z = 0.f;
    for (int l = 0; l < 32; l++) { float p = __expf(sc[t][l] - m); sc[t][l] = p; z += p; }
    float zi = 1.f / z;
    for (int l = 0; l < 32; l++) sc[t][l] *= zi;
  }
  __syncthreads();
  const int os = t >> 3, oe = (t & 7) * 8;
  float o[8];
#pragma unroll
  for (int j = 0; j < 8; j++) o[j] = 0.f;
  for (int l = 0; l < 32; l++) {
    float p = sc[os][l];
#pragma unroll
    for (int j = 0; j < 8; j++) o[j] += p * vv[l][oe + j];
  }
#pragma unroll
  for (int j = 0; j < 8; j++)
    Obf[((size_t)(b * 32 + os)) * 256 + h * 64 + oe + j] = (__bf16)o[j];
}

// ---------------------------------------------------------------------------
// attn_3: single query pools 32 tokens; one block (64 threads) per (b,h)
// ---------------------------------------------------------------------------
__global__ void k_attn3(const float* __restrict__ K3, const float* __restrict__ V3,
                        const float* __restrict__ q3, __bf16* __restrict__ Obf) {
  __shared__ float sc[32];
  const int b = blockIdx.x >> 2, h = blockIdx.x & 3;
  const int t = threadIdx.x;
  if (t < 32) {
    float a = 0.f;
#pragma unroll 8
    for (int e = 0; e < 64; e++)
      a += q3[h * 64 + e] * 0.125f * K3[((size_t)(b * 32 + t)) * 256 + h * 64 + e];
    sc[t] = a;
  }
  __syncthreads();
  if (t == 0) {
    float m = -1e30f;
    for (int l = 0; l < 32; l++) m = fmaxf(m, sc[l]);
    float z = 0.f;
    for (int l = 0; l < 32; l++) { float p = __expf(sc[l] - m); sc[l] = p; z += p; }
    float zi = 1.f / z;
    for (int l = 0; l < 32; l++) sc[l] *= zi;
  }
  __syncthreads();
  float o = 0.f;
  for (int l = 0; l < 32; l++) o += sc[l] * V3[((size_t)(b * 32 + l)) * 256 + h * 64 + t];
  Obf[(size_t)b * 256 + h * 64 + t] = (__bf16)o;
}

// q3 = seed3 @ qw3   (tiny 1x256 x 256x256)
__global__ void k_vec_mat(const float* __restrict__ v, const float* __restrict__ W,
                          float* __restrict__ out) {
  int c = threadIdx.x;
  float a = 0.f;
  for (int d = 0; d < 256; d++) a += v[d] * W[(size_t)d * 256 + c];
  out[c] = a;
}

// ---------------------------------------------------------------------------
// Host launch
// ---------------------------------------------------------------------------
extern "C" void kernel_launch(void* const* d_in, const int* in_sizes, int n_in,
                              void* d_out, int out_size, void* d_ws, size_t ws_size,
                              hipStream_t stream) {
  const float* x       = (const float*)d_in[0];
  const float* conv_W0 = (const float*)d_in[1];
  const float* conv_b0 = (const float*)d_in[2];
  const float* conv_W1 = (const float*)d_in[3];
  const float* conv_b1 = (const float*)d_in[4];
  const float* conv_W2 = (const float*)d_in[5];
  const float* conv_b2 = (const float*)d_in[6];
  const float* seed1   = (const float*)d_in[7];
  const float* Wk1     = (const float*)d_in[8];
  const float* bk1     = (const float*)d_in[9];
  const float* Wv1     = (const float*)d_in[10];
  const float* bv1     = (const float*)d_in[11];
  const float* qw1     = (const float*)d_in[12];
  const float* ow1     = (const float*)d_in[13];
  const float* ob1     = (const float*)d_in[14];
  const float* lnh1_w  = (const float*)d_in[15];
  const float* lnh1_b  = (const float*)d_in[16];
  const float* lnz1_w  = (const float*)d_in[17];
  const float* lnz1_b  = (const float*)d_in[18];
  const float* ff1_W   = (const float*)d_in[19];
  const float* ff1_b   = (const float*)d_in[20];
  const float* in_w2   = (const float*)d_in[21];
  const float* in_b2   = (const float*)d_in[22];
  const float* ow2     = (const float*)d_in[23];
  const float* ob2     = (const float*)d_in[24];
  const float* lnh2_w  = (const float*)d_in[25];
  const float* lnh2_b  = (const float*)d_in[26];
  const float* lnz2_w  = (const float*)d_in[27];
  const float* lnz2_b  = (const float*)d_in[28];
  const float* ff2_W   = (const float*)d_in[29];
  const float* ff2_b   = (const float*)d_in[30];
  const float* seed3   = (const float*)d_in[31];
  const float* Wk3     = (const float*)d_in[32];
  const float* bk3     = (const float*)d_in[33];
  const float* Wv3     = (const float*)d_in[34];
  const float* bv3     = (const float*)d_in[35];
  const float* qw3     = (const float*)d_in[36];
  const float* ow3     = (const float*)d_in[37];
  const float* ob3     = (const float*)d_in[38];
  const float* lnh3_w  = (const float*)d_in[39];
  const float* lnh3_b  = (const float*)d_in[40];
  const float* lnz3_w  = (const float*)d_in[41];
  const float* lnz3_b  = (const float*)d_in[42];
  const float* ff3_W   = (const float*)d_in[43];
  const float* ff3_b   = (const float*)d_in[44];
  const float* hW1     = (const float*)d_in[45];
  const float* hb1     = (const float*)d_in[46];
  const float* hW2     = (const float*)d_in[47];
  const float* hb2     = (const float*)d_in[48];
  const int*   srcp    = (const int*)d_in[49];
  const int*   dstp    = (const int*)d_in[50];

  char* ws = (char*)d_ws;
  float*  bufA  = (float*)(ws + O_A);
  float*  bufB  = (float*)(ws + O_Bg);
  float*  bufC  = (float*)(ws + O_Cg);
  __bf16* XB    = (__bf16*)(ws + O_XB);
  float*  dinv  = (float*)(ws + O_DINV);
  float*  nrm   = (float*)(ws + O_NORM);
  __bf16* w0t   = (__bf16*)(ws + O_W0T);
  __bf16* w1t   = (__bf16*)(ws + O_W1T);
  __bf16* w2t   = (__bf16*)(ws + O_W2T);
  __bf16* wk1t  = (__bf16*)(ws + O_WK1T);
  __bf16* wv1t  = (__bf16*)(ws + O_WV1T);
  __bf16* qw1t  = (__bf16*)(ws + O_QW1T);
  __bf16* ow1t  = (__bf16*)(ws + O_OW1T);
  __bf16* ff1t  = (__bf16*)(ws + O_FF1T);
  __bf16* inw2t = (__bf16*)(ws + O_INW2T);
  __bf16* ow2t  = (__bf16*)(ws + O_OW2T);
  __bf16* ff2t  = (__bf16*)(ws + O_FF2T);
  __bf16* wk3t  = (__bf16*)(ws + O_WK3T);
  __bf16* wv3t  = (__bf16*)(ws + O_WV3T);
  __bf16* ow3t  = (__bf16*)(ws + O_OW3T);
  __bf16* ff3t  = (__bf16*)(ws + O_FF3T);
  __bf16* hw1t  = (__bf16*)(ws + O_HW1T);
  __bf16* hw2t  = (__bf16*)(ws + O_HW2T);
  __bf16* seedb = (__bf16*)(ws + O_SEEDB);
  float*  QP1   = (float*)(ws + O_QP1);
  __bf16* OBF   = (__bf16*)(ws + O_OBF);
  float*  ATT   = (float*)(ws + O_ATT);
  float*  Ebuf  = (float*)(ws + O_E);
  float*  X1    = (float*)(ws + O_X1);
  float*  FFb   = (float*)(ws + O_FF);
  float*  X2    = (float*)(ws + O_X2);
  __bf16* SB    = (__bf16*)(ws + O_SB);
  float*  QKV   = (float*)(ws + O_QKV);
  float*  K3    = (float*)(ws + O_K3);
  float*  V3    = (float*)(ws + O_V3);
  float*  Q3    = (float*)(ws + O_Q3);
  float*  yout  = (float*)d_out;

  auto cdiv = [](int a, int b) { return (a + b - 1) / b; };
  auto gemm = [&](const __bf16* A_, const __bf16* Bt_, const float* bias_,
                  float* C_, int M_, int N_, int K_, int relu_) {
    int waves = (M_ / 16) * (N_ / 16);
    k_gemm_wmma<<<cdiv(waves, 8), 256, 0, stream>>>(A_, Bt_, bias_, C_, M_, N_, K_, relu_);
  };
  auto tobf = [&](const float* s, __bf16* d, int n, int relu) {
    k_f32_to_bf16<<<cdiv(n, 256), 256, 0, stream>>>(s, d, n, relu);
  };
  auto wT = [&](const float* W, __bf16* Wt, int K_, int N_) {
    k_w_t_bf16<<<cdiv(K_ * N_, 256), 256, 0, stream>>>(W, Wt, K_, N_);
  };

  // --- weights -> bf16 transposed ---
  wT(conv_W0, w0t, 128, 256);
  wT(conv_W1, w1t, 256, 256);
  wT(conv_W2, w2t, 256, 256);
  k_wkcat_t<<<cdiv(256 * 256, 256), 256, 0, stream>>>(Wk1, wk1t);
  k_wkcat_t<<<cdiv(256 * 256, 256), 256, 0, stream>>>(Wv1, wv1t);
  wT(qw1, qw1t, 256, 256);
  wT(ow1, ow1t, 256, 256);
  wT(ff1_W, ff1t, 256, 256);
  wT(in_w2, inw2t, 256, 768);
  wT(ow2, ow2t, 256, 256);
  wT(ff2_W, ff2t, 256, 256);
  wT(Wk3, wk3t, 256, 256);
  wT(Wv3, wv3t, 256, 256);
  wT(ow3, ow3t, 256, 256);
  wT(ff3_W, ff3t, 256, 256);
  wT(hW1, hw1t, 256, 256);
  wT(hW2, hw2t, 256, 128);
  tobf(seed1, seedb, 32 * 256, 0);

  // --- GCN normalization ---
  k_fill<<<cdiv(NTOT, 256), 256, 0, stream>>>(dinv, 1.0f, NTOT);   // self-loop
  k_deg<<<cdiv(Ec, 256), 256, 0, stream>>>(dstp, dinv, Ec);
  k_dinv<<<cdiv(NTOT, 256), 256, 0, stream>>>(dinv, NTOT);
  k_edge_norm<<<cdiv(Ec, 256), 256, 0, stream>>>(srcp, dstp, dinv, nrm, Ec);

  // --- 3 GCN layers (proj -> init(self+bias) -> edge scatter -> relu/bf16) ---
  tobf(x, XB, NTOT * DINc, 0);
  const __bf16* Wts[3] = {w0t, w1t, w2t};
  const float* bs[3] = {conv_b0, conv_b1, conv_b2};
  int Ks[3] = {128, 256, 256};
  for (int l = 0; l < 3; l++) {
    gemm(XB, Wts[l], nullptr, bufA, NTOT, 256, Ks[l], 0);
    k_gcn_init<<<cdiv(NTOT * Dm, 256), 256, 0, stream>>>(bufA, bs[l], dinv, bufB);
    k_gcn_scatter<<<cdiv(Ec * 64, 256), 256, 0, stream>>>(bufA, srcp, dstp, nrm, bufB);
    tobf(bufB, XB, NTOT * Dm, 1);
  }

  // --- K/V head-convs (bias BEFORE aggregation, no post-bias) ---
  gemm(XB, wk1t, bk1, bufA, NTOT, 256, 256, 0);
  k_gcn_init<<<cdiv(NTOT * Dm, 256), 256, 0, stream>>>(bufA, nullptr, dinv, bufB);
  k_gcn_scatter<<<cdiv(Ec * 64, 256), 256, 0, stream>>>(bufA, srcp, dstp, nrm, bufB);
  gemm(XB, wv1t, bv1, bufA, NTOT, 256, 256, 0);
  k_gcn_init<<<cdiv(NTOT * Dm, 256), 256, 0, stream>>>(bufA, nullptr, dinv, bufC);
  k_gcn_scatter<<<cdiv(Ec * 64, 256), 256, 0, stream>>>(bufA, srcp, dstp, nrm, bufC);

  // --- attn_1 ---
  gemm(seedb, qw1t, nullptr, QP1, 32, 256, 256, 0);         // q = seed1 @ qw1
  k_attn1<<<Bc * Hc, 256, 0, stream>>>(bufB, bufC, QP1, OBF);
  gemm(OBF, ow1t, ob1, ATT, 2048, 256, 256, 0);             // out-proj [s*B+b, D]
  k_ln_add_bcast<<<cdiv(2048 * 32, 256), 256, 0, stream>>>(seed1, ATT, lnh1_w, lnh1_b, Ebuf, 32, 64);
  tobf(Ebuf, SB, 2048 * 256, 0);
  gemm(SB, ff1t, ff1_b, FFb, 2048, 256, 256, 0);
  k_ln_add<<<cdiv(2048 * 32, 256), 256, 0, stream>>>(Ebuf, FFb, lnz1_w, lnz1_b, X1, 2048);

  // --- attn_2 (self-MHA over 32 tokens) ---
  tobf(X1, SB, 2048 * 256, 0);
  gemm(SB, inw2t, in_b2, QKV, 2048, 768, 256, 0);
  k_attn2<<<Bc * Hc, 256, 0, stream>>>(QKV, OBF);
  gemm(OBF, ow2t, ob2, ATT, 2048, 256, 256, 0);
  k_ln_add<<<cdiv(2048 * 32, 256), 256, 0, stream>>>(X1, ATT, lnh2_w, lnh2_b, Ebuf, 2048);
  tobf(Ebuf, SB, 2048 * 256, 0);
  gemm(SB, ff2t, ff2_b, FFb, 2048, 256, 256, 0);
  k_ln_add<<<cdiv(2048 * 32, 256), 256, 0, stream>>>(Ebuf, FFb, lnz2_w, lnz2_b, X2, 2048);

  // --- attn_3 (1 seed pools 32 tokens) ---
  tobf(X2, SB, 2048 * 256, 0);
  gemm(SB, wk3t, bk3, K3, 2048, 256, 256, 0);
  gemm(SB, wv3t, bv3, V3, 2048, 256, 256, 0);
  k_vec_mat<<<1, 256, 0, stream>>>(seed3, qw3, Q3);
  k_attn3<<<Bc * Hc, 64, 0, stream>>>(K3, V3, Q3, OBF);     // uses first 64x256 of OBF
  gemm(OBF, ow3t, ob3, ATT, 64, 256, 256, 0);
  k_ln_add_bcast<<<cdiv(64 * 32, 256), 256, 0, stream>>>(seed3, ATT, lnh3_w, lnh3_b, Ebuf, 1, 64);
  tobf(Ebuf, SB, 64 * 256, 0);
  gemm(SB, ff3t, ff3_b, FFb, 64, 256, 256, 0);
  k_ln_add<<<cdiv(64 * 32, 256), 256, 0, stream>>>(Ebuf, FFb, lnz3_w, lnz3_b, X1, 64); // x3

  // --- head: y = (x3 @ hW1 + hb1) @ hW2 + hb2 ---
  tobf(X1, SB, 64 * 256, 0);
  gemm(SB, hw1t, hb1, FFb, 64, 256, 256, 0);
  tobf(FFb, SB, 64 * 256, 0);
  gemm(SB, hw2t, hb2, yout, 64, 128, 256, 0);
  (void)in_sizes; (void)n_in; (void)out_size; (void)ws_size;
}
